// LSTMCellQ_bmm_28020366639341
// MI455X (gfx1250) — matrix operand order repaired and checked by reading, not compiled, not run on hardware
//
#include <hip/hip_runtime.h>

// ---------------------------------------------------------------------------
// Quantized LSTM cell for MI455X (gfx1250), wave32 + WMMA bf16 path.
//   pass 1: fuse clip/quant/noise -> bf16 effective weights, LDS tile-transpose
//           to [n][col(1024)][k(256)] with coalesced reads AND writes
//   pass 2: pact+quant activations -> bf16
//   pass 3: quant biases + noise -> f32
//   pass 4: dual WMMA GEMM (x@Wih, h@Whh) interleaved + fused gate elementwise,
//           A operands staged via async global->LDS (ASYNCcnt path)
// ---------------------------------------------------------------------------

typedef __attribute__((ext_vector_type(16))) __bf16 v16bf;
typedef __attribute__((ext_vector_type(8)))  float  v8f;
typedef __attribute__((ext_vector_type(4)))  int    v4i_t;

#define NBLK 48
#define ISZ  256
#define HID  256
#define BATCH 256
#define H4   1024

#if __has_builtin(__builtin_amdgcn_global_load_async_to_lds_b128) && \
    __has_builtin(__builtin_amdgcn_s_wait_asynccnt)
#define USE_ASYNC_LDS 1
typedef __attribute__((address_space(1))) v4i_t g_v4i;
typedef __attribute__((address_space(3))) v4i_t l_v4i;
#else
#define USE_ASYNC_LDS 0
#endif

union FragBF { v16bf v; uint4 q[2]; };

__device__ __forceinline__ unsigned short f2bf(float f) {
    unsigned int u = __float_as_uint(f);
    u += 0x7FFFu + ((u >> 16) & 1u);          // round-to-nearest-even
    return (unsigned short)(u >> 16);
}
__device__ __forceinline__ float pactf(float x, float a) {
    return fminf(fmaxf(x, -a), a);
}
__device__ __forceinline__ float quant4(float x) {           // WB=4, range 1.0
    float xc = fminf(fmaxf(x, -0.875f), 0.875f);
    return rintf(xc * 8.0f) * 0.125f;
}
__device__ __forceinline__ float sigm(float x) {
    return 1.0f / (1.0f + __expf(-x));
}

// ---- pass 1: effective weights, bf16, tile-transposed to [n][c][k] --------
// grid = (16 c-tiles, 4 k-tiles, 48 blocks), 256 threads. 64x64 tiles via LDS
// so the strided transpose never touches HBM: f32 reads coalesced along c,
// bf16 writes coalesced along k.
__global__ __launch_bounds__(256) void prep_weights(
    const float* __restrict__ w_ih, const float* __restrict__ n_ih,
    const float* __restrict__ w_hh, const float* __restrict__ n_hh,
    const float* __restrict__ mask,
    unsigned short* __restrict__ W1t, unsigned short* __restrict__ W2t) {
    __shared__ unsigned short t1[64][65];
    __shared__ unsigned short t2[64][65];
    const int n  = blockIdx.z;
    const int k0 = blockIdx.y * 64;
    const int c0 = blockIdx.x * 64;
    const int row = threadIdx.x >> 6;    // 0..3
    const int col = threadIdx.x & 63;    // 0..63

    const size_t base = ((size_t)n * ISZ + k0) * H4 + c0;
#pragma unroll
    for (int r = 0; r < 16; ++r) {
        const int kk = r * 4 + row;
        const size_t gi = base + (size_t)kk * H4 + col;
        float v1 = quant4(fminf(fmaxf(w_ih[gi], -1.f), 1.f)) + n_ih[gi];
        float v2 = quant4(fminf(fmaxf(w_hh[gi] * mask[gi], -1.f), 1.f)) + n_hh[gi];
        t1[kk][col] = f2bf(v1);
        t2[kk][col] = f2bf(v2);
    }
    __syncthreads();

    const size_t obase = ((size_t)n * H4 + c0) * 256 + k0;
#pragma unroll
    for (int r = 0; r < 16; ++r) {
        const int cc = r * 4 + row;
        const size_t oi = obase + (size_t)cc * 256 + col;   // contiguous in col
        W1t[oi] = t1[col][cc];
        W2t[oi] = t2[col][cc];
    }
}

// ---- pass 2: quantized activations, bf16, layout [n][b][k] -----------------
__global__ __launch_bounds__(256) void prep_acts(
    const float* __restrict__ x, const float* __restrict__ h,
    const float* __restrict__ a1, const float* __restrict__ a11,
    unsigned short* __restrict__ Xq, unsigned short* __restrict__ Hq) {
    size_t i = (size_t)blockIdx.x * 256 + threadIdx.x;       // 48*256*256
    int n = (int)(i >> 16);
    float ax = a1[n], ah = a11[n];
    const float lim = 127.0f / 128.0f;
    float px = pactf(x[i], ax);
    float tx = fminf(fmaxf(px / ax, -lim), lim);
    Xq[i] = f2bf(rintf(tx * 128.0f) * (ax * (1.0f / 128.0f)));
    float ph = pactf(h[i], ah);
    float th = fminf(fmaxf(ph / ah, -lim), lim);
    Hq[i] = f2bf(rintf(th * 128.0f) * (ah * (1.0f / 128.0f)));
}

// ---- pass 3: biases -------------------------------------------------------
__global__ __launch_bounds__(256) void prep_bias(
    const float* __restrict__ b_ih, const float* __restrict__ nb_ih,
    const float* __restrict__ b_hh, const float* __restrict__ nb_hh,
    float* __restrict__ B1, float* __restrict__ B2) {
    int i = blockIdx.x * 256 + threadIdx.x;                  // 48*1024
    B1[i] = quant4(fminf(fmaxf(b_ih[i], -2.f), 2.f)) + nb_ih[i];
    B2[i] = quant4(fminf(fmaxf(b_hh[i], -2.f), 2.f)) + nb_hh[i];
}

// ---- pass 4: dual WMMA GEMM + fused LSTM gates ----------------------------
// grid = (4 col tiles, 4 row tiles, 48 blocks), 256 threads = 8 waves.
// WG tile: 64 batch rows x 64 hidden cols. Wave (rt, half) owns rows rt*16..+15
// and column subtiles cs in {2*half, 2*half+1}; for each cs it computes all
// four gate tiles (i,j,f,o) so the elementwise combine is register-local.
__global__ __launch_bounds__(256) void lstm_wmma(
    const unsigned short* __restrict__ Xq, const unsigned short* __restrict__ Hq,
    const unsigned short* __restrict__ W1t, const unsigned short* __restrict__ W2t,
    const float* __restrict__ B1, const float* __restrict__ B2,
    const float* __restrict__ cx,
    const float* __restrict__ a3,  const float* __restrict__ a4,
    const float* __restrict__ a5,  const float* __restrict__ a6,
    const float* __restrict__ a7,  const float* __restrict__ a8,
    const float* __restrict__ a9,  const float* __restrict__ a10,
    const float* __restrict__ a11, const float* __restrict__ a12,
    const float* __restrict__ a13, const float* __restrict__ a14,
    float* __restrict__ outH, float* __restrict__ outC) {

    __shared__ __align__(16) unsigned short sX[64 * 256];
    __shared__ __align__(16) unsigned short sH[64 * 256];

    const int n  = blockIdx.z;
    const int c0 = blockIdx.x * 64;
    const int r0 = blockIdx.y * 64;
    const int tid = threadIdx.x;

    // Stage A operands (64 rows x 256 K, bf16) into LDS: contiguous copy.
    {
#if USE_ASYNC_LDS
        // async global->LDS (tracked by ASYNCcnt), bypasses VGPR round-trip
        g_v4i* ggx = (g_v4i*)(Xq + ((size_t)n * BATCH + r0) * 256);
        g_v4i* ggh = (g_v4i*)(Hq + ((size_t)n * BATCH + r0) * 256);
        l_v4i* dx  = (l_v4i*)sX;
        l_v4i* dh  = (l_v4i*)sH;
#pragma unroll
        for (int t = tid; t < 2048; t += 256) {
            __builtin_amdgcn_global_load_async_to_lds_b128(ggx + t, dx + t, 0, 0);
            __builtin_amdgcn_global_load_async_to_lds_b128(ggh + t, dh + t, 0, 0);
        }
        __builtin_amdgcn_s_wait_asynccnt(0);
#else
        const uint4* gx = (const uint4*)(Xq + ((size_t)n * BATCH + r0) * 256);
        const uint4* gh = (const uint4*)(Hq + ((size_t)n * BATCH + r0) * 256);
        uint4* dx = (uint4*)sX;
        uint4* dh = (uint4*)sH;
#pragma unroll
        for (int t = tid; t < 2048; t += 256) { dx[t] = gx[t]; dh[t] = gh[t]; }
#endif
    }
    __syncthreads();

    const int wave  = tid >> 5;
    const int lane  = tid & 31;
    const int rt    = wave >> 1;      // row subtile 0..3
    const int chalf = wave & 1;       // column half
    const int lhi   = lane >> 4;      // 0 | 1
    const int lcol  = lane & 15;

    const unsigned short* sArow = sX + (rt * 16 + lcol) * 256;
    const unsigned short* sHrow = sH + (rt * 16 + lcol) * 256;

    const float sA3 = a3[n],  sA4 = a4[n],  sA5 = a5[n],  sA6 = a6[n];
    const float sA7 = a7[n],  sA8 = a8[n],  sA9 = a9[n],  sA10 = a10[n];
    const float sA11 = a11[n], sA12 = a12[n], sA13 = a13[n], sA14 = a14[n];

    for (int ci = 0; ci < 2; ++ci) {
        const int cs  = chalf * 2 + ci;
        const int col = c0 + cs * 16 + lcol;    // hidden column of this lane
        float G[4][8];

#pragma unroll
        for (int g = 0; g < 4; ++g) {
            const int gc = g * 256 + col;       // gate column in [0,1024)
            const unsigned short* pB1 = W1t + ((size_t)n * H4 + gc) * 256;
            const unsigned short* pB2 = W2t + ((size_t)n * H4 + gc) * 256;
            if (g < 3) {                        // prefetch next gate's B rows
                __builtin_prefetch(pB1 + 256 * 256, 0, 1);
                __builtin_prefetch(pB2 + 256 * 256, 0, 1);
            }

            v8f acc1 = {0.f, 0.f, 0.f, 0.f, 0.f, 0.f, 0.f, 0.f};
            v8f acc2 = {0.f, 0.f, 0.f, 0.f, 0.f, 0.f, 0.f, 0.f};
            // Interleave the two GEMMs: two independent WMMA chains so B-load
            // latency of one chain hides under the other chain's WMMA.
#pragma unroll
            for (int ks = 0; ks < 8; ++ks) {
                const int kk = ks * 32;
                const int kb = kk + 16 * lhi;
                FragBF a1f, b1f, a2f, b2f;
                // A 16x32 bf16: lanes 0-15 -> K = kk+0..7 / kk+16..23,
                //               lanes16-31 -> K = kk+8..15 / kk+24..31
                a1f.q[0] = *(const uint4*)(sArow + kk + 8 * lhi);
                a1f.q[1] = *(const uint4*)(sArow + kk + 16 + 8 * lhi);
                a2f.q[0] = *(const uint4*)(sHrow + kk + 8 * lhi);
                a2f.q[1] = *(const uint4*)(sHrow + kk + 16 + 8 * lhi);
                // B 32x16 bf16 (weights pre-transposed => contiguous K)
                b1f.q[0] = *(const uint4*)(pB1 + kk + 8 * lhi);
                b1f.q[1] = *(const uint4*)(pB1 + kk + 16 + 8 * lhi);
                b2f.q[0] = *(const uint4*)(pB2 + kk + 8 * lhi);
                b2f.q[1] = *(const uint4*)(pB2 + kk + 16 + 8 * lhi);
                acc1 = __builtin_amdgcn_wmma_f32_16x16x32_bf16(
                    false, a1f.v, false, b1f.v, (short)0, acc1, false, false);
                acc2 = __builtin_amdgcn_wmma_f32_16x16x32_bf16(
                    false, a2f.v, false, b2f.v, (short)0, acc2, false, false);
            }

            const float b1 = B1[(size_t)n * H4 + gc];
            const float b2 = B2[(size_t)n * H4 + gc];
#pragma unroll
            for (int e = 0; e < 8; ++e) {
                G[g][e] = pactf(pactf(acc1[e] + b1, sA12) +
                                pactf(acc2[e] + b2, sA13), sA14);
            }
        }

        // Elementwise LSTM gate math; C/D layout: row = e + 8*lhi, col = lcol.
#pragma unroll
        for (int e = 0; e < 8; ++e) {
            const int row = r0 + rt * 16 + e + 8 * lhi;
            const size_t oidx = ((size_t)n * BATCH + row) * HID + col;
            const float cv = cx[oidx];
            const float fg   = pactf(sigm(G[2][e]), sA3);
            const float ig   = pactf(sigm(G[0][e]), sA4);
            const float act  = pactf(tanhf(G[1][e]), sA5);
            const float og   = pactf(sigm(G[3][e]), sA6);
            const float gcll = pactf(cv * fg, sA7);
            const float ain  = pactf(ig * act, sA8);
            const float nc   = pactf(gcll + ain, sA9);
            const float acll = pactf(tanhf(nc), sA10);
            const float nh   = pactf(acll * og, sA11);
            outH[oidx] = nh;
            outC[oidx] = nc;
        }
    }
}

// ---------------------------------------------------------------------------
extern "C" void kernel_launch(void* const* d_in, const int* in_sizes, int n_in,
                              void* d_out, int out_size, void* d_ws, size_t ws_size,
                              hipStream_t stream) {
    const float* input  = (const float*)d_in[0];
    const float* hx     = (const float*)d_in[1];
    const float* cx     = (const float*)d_in[2];
    const float* w_mask = (const float*)d_in[3];
    const float* w_ih   = (const float*)d_in[4];
    const float* w_hh   = (const float*)d_in[5];
    const float* b_ih   = (const float*)d_in[6];
    const float* b_hh   = (const float*)d_in[7];
    const float* n_ih_w = (const float*)d_in[8];
    const float* n_ih_b = (const float*)d_in[9];
    const float* n_hh_w = (const float*)d_in[10];
    const float* n_hh_b = (const float*)d_in[11];
    const float* a1  = (const float*)d_in[12];
    const float* a3  = (const float*)d_in[13];
    const float* a4  = (const float*)d_in[14];
    const float* a5  = (const float*)d_in[15];
    const float* a6  = (const float*)d_in[16];
    const float* a7  = (const float*)d_in[17];
    const float* a8  = (const float*)d_in[18];
    const float* a9  = (const float*)d_in[19];
    const float* a10 = (const float*)d_in[20];
    const float* a11 = (const float*)d_in[21];
    const float* a12 = (const float*)d_in[22];
    const float* a13 = (const float*)d_in[23];
    const float* a14 = (const float*)d_in[24];

    const size_t ACT = (size_t)NBLK * BATCH * 256;   // 3,145,728 bf16 elems
    const size_t WQ  = (size_t)NBLK * H4 * 256;      // 12,582,912 bf16 elems

    unsigned short* Xq  = (unsigned short*)d_ws;
    unsigned short* Hq  = Xq + ACT;
    unsigned short* W1t = Hq + ACT;
    unsigned short* W2t = W1t + WQ;
    float* B1 = (float*)(W2t + WQ);
    float* B2 = B1 + (size_t)NBLK * H4;

    float* outH = (float*)d_out;
    float* outC = outH + (size_t)NBLK * BATCH * HID;

    prep_weights<<<dim3(16, 4, NBLK), 256, 0, stream>>>(
        w_ih, n_ih_w, w_hh, n_hh_w, w_mask, W1t, W2t);
    prep_acts<<<(unsigned)(ACT / 256), 256, 0, stream>>>(
        input, hx, a1, a11, Xq, Hq);
    prep_bias<<<(NBLK * H4) / 256, 256, 0, stream>>>(
        b_ih, n_ih_b, b_hh, n_hh_b, B1, B2);
    lstm_wmma<<<dim3(4, 4, NBLK), 256, 0, stream>>>(
        Xq, Hq, W1t, W2t, B1, B2, cx,
        a3, a4, a5, a6, a7, a8, a9, a10, a11, a12, a13, a14,
        outH, outC);
}
